// MobileRaftNet_76020921140012
// MI455X (gfx1250) — matrix operand order repaired and checked by reading, not compiled, not run on hardware
//
#include <hip/hip_runtime.h>
#include <hip/hip_bf16.h>
#include <math.h>

typedef __attribute__((ext_vector_type(16))) __bf16        v16bf;
typedef __attribute__((ext_vector_type(8)))  float         v8f;
typedef __attribute__((ext_vector_type(4)))  float         v4f;
typedef __attribute__((ext_vector_type(4)))  unsigned int  v4ui;

#define N_    4
#define C_    128
#define H_    96
#define W_    160
#define HW_   (H_*W_)
#define NHW_  ((size_t)N_*H_*W_)
#define NCHW_ ((size_t)N_*C_*H_*W_)
#define WT_   (W_/16)          // 10 width tiles (corr kernel)
#define WT2_  (W_/32)          // 5 width tiles  (conv kernel, M=32 per block)

union Frag { v16bf v; v4ui u[2]; };

__device__ inline unsigned pack2bf(float x, float y) {
  union { __bf16 h; unsigned short s; } a, b;
  a.h = (__bf16)x; b.h = (__bf16)y;
  return ((unsigned)b.s << 16) | (unsigned)a.s;
}

// ---------------------------------------------------------------- utilities
__global__ void k_f32_to_bf16(const float* __restrict__ src, __bf16* __restrict__ dst, size_t n) {
  size_t i = (size_t)blockIdx.x * 256 + threadIdx.x;
  if (i < n) dst[i] = (__bf16)src[i];
}

__global__ void k_zero_f32(float* __restrict__ dst, size_t n) {
  size_t i = (size_t)blockIdx.x * 256 + threadIdx.x;
  if (i < n) dst[i] = 0.0f;
}

__global__ void k_hprep(const float* __restrict__ h0, float* __restrict__ hf,
                        __bf16* __restrict__ hb, size_t n) {
  size_t i = (size_t)blockIdx.x * 256 + threadIdx.x;
  if (i < n) { float v = h0[i]; hf[i] = v; hb[i] = (__bf16)v; }
}

// pack f32 weights [Cout][CinSrc][KK] -> bf16 [CoutPad][K = tap*CinDst + c] (tap-major)
__global__ void k_pack_w(const float* __restrict__ src, __bf16* __restrict__ dst,
                         int Cout, int CoutPad, int CinSrc, int CinDst, int KK) {
  size_t i = (size_t)blockIdx.x * 256 + threadIdx.x;
  int Kd = CinDst * KK;
  size_t total = (size_t)CoutPad * Kd;
  if (i >= total) return;
  int co = (int)(i / Kd), k = (int)(i % Kd);
  int c = k % CinDst, tap = k / CinDst;
  float v = (co < Cout && c < CinSrc) ? src[((size_t)co * CinSrc + c) * KK + tap] : 0.0f;
  dst[i] = (__bf16)v;
}

// ---------------------------------------------------------------- correlation
// pyr0[n][h][w][v] = sum_c l[n,c,h,w]*r[n,c,h,v]; per (n,h): [160x128]*[128x160]
__global__ __launch_bounds__(320)
void k_corr_wmma(const __bf16* __restrict__ l, const __bf16* __restrict__ r,
                 float* __restrict__ pyr0) {
  __shared__ __align__(16) __bf16 At[16 * 136];     // [m][k], padded stride
  __shared__ __align__(16) __bf16 Bt[160 * 136];    // [v][k], padded stride
  const int tid = threadIdx.x;
  const int wt = blockIdx.x % WT_;
  const int nh = blockIdx.x / WT_;
  const int hh = nh % H_, nn = nh / H_;
  const int w0 = wt * 16;
  for (int idx = tid; idx < 16 * 128; idx += 320) {
    int m = idx >> 7, k = idx & 127;
    At[m * 136 + k] = l[(((size_t)nn * C_ + k) * H_ + hh) * W_ + w0 + m];
  }
  for (int idx = tid; idx < 160 * 128; idx += 320) {
    int k = idx / 160, v = idx % 160;               // reads coalesced along v
    Bt[v * 136 + k] = r[(((size_t)nn * C_ + k) * H_ + hh) * W_ + v];
  }
  __syncthreads();
  const int lane = tid & 31, wave = tid >> 5;
  const int vbase = wave * 16;
  const int m = lane & 15, agrp = (lane >> 4) * 8;
  const int nl = lane & 15, bgrp = (lane >> 4) * 8;
  const __bf16* arow = At + m * 136;
  const __bf16* brow = Bt + (vbase + nl) * 136;
  v8f acc = {};
#pragma unroll
  for (int kb = 0; kb < 128; kb += 32) {
    Frag a, b;
    a.u[0] = *(const v4ui*)(arow + kb + agrp);
    a.u[1] = *(const v4ui*)(arow + kb + 16 + agrp);
    b.u[0] = *(const v4ui*)(brow + kb + bgrp);
    b.u[1] = *(const v4ui*)(brow + kb + bgrp + 16);
    acc = __builtin_amdgcn_wmma_f32_16x16x32_bf16(false, a.v, false, b.v, (short)0, acc,
                                                  false, false);
  }
  const int mrow = (lane >> 4) * 8;
#pragma unroll
  for (int rr = 0; rr < 8; ++rr) {
    int M = mrow + rr;
    pyr0[((((size_t)nn * H_ + hh) * W_) + (w0 + M)) * W_ + vbase + nl] = acc[rr];
  }
}

// average pool pairs along last dim: in [rows][Win] -> out [rows][Win/2]
__global__ void k_pool_half(const float* __restrict__ in, float* __restrict__ out,
                            int Win, size_t rows) {
  size_t i = (size_t)blockIdx.x * 256 + threadIdx.x;
  int Wo = Win >> 1;
  size_t total = rows * (size_t)Wo;
  if (i >= total) return;
  size_t row = i / Wo; int c = (int)(i % Wo);
  out[i] = 0.5f * (in[row * Win + 2 * c] + in[row * Win + 2 * c + 1]);
}

// sample corr pyramid -> corrf bf16 [N][64][H][W] (36 real channels, rest zero)
__global__ void k_sample(const float* __restrict__ p0, const float* __restrict__ p1,
                         const float* __restrict__ p2, const float* __restrict__ p3,
                         const float* __restrict__ flow, __bf16* __restrict__ corrf) {
  size_t i = (size_t)blockIdx.x * 256 + threadIdx.x;
  size_t total = (size_t)N_ * 64 * HW_;
  if (i >= total) return;
  int hw = (int)(i % HW_);
  int ch = (int)((i / HW_) % 64);
  int nn = (int)(i / ((size_t)64 * HW_));
  if (ch >= 36) { corrf[i] = (__bf16)0.0f; return; }
  int lvl = ch / 9, j = ch % 9;
  int hh = hw / W_, ww = hw % W_;
  float x0 = (float)ww + flow[(size_t)nn * HW_ + hw];
  float xs = x0 / (float)(1 << lvl) + (float)(j - 4);
  float xf = floorf(xs);
  float fr = xs - xf;
  int i0 = (int)xf, i1 = i0 + 1;
  int Wl = W_ >> lvl;
  const float* p = (lvl == 0) ? p0 : (lvl == 1) ? p1 : (lvl == 2) ? p2 : p3;
  size_t base = (((size_t)nn * H_ + hh) * W_ + ww) * Wl;
  float v0 = (i0 >= 0 && i0 < Wl) ? p[base + i0] : 0.0f;
  float v1 = (i1 >= 0 && i1 < Wl) ? p[base + i1] : 0.0f;
  corrf[i] = (__bf16)(v0 * (1.0f - fr) + v1 * fr);
}

// ---------------------------------------------------------------- WMMA conv
// implicit-GEMM conv, NCHW bf16 in, weights bf16 [CoutPad][K] with K tap-major.
// Block computes a 32(spatial) x 128(cout) tile: 8 waves, each wave owns 16
// couts and two 16-wide spatial sub-tiles that share one B (weight) fragment
// -> 2 WMMAs per K-step per wave, half the weight traffic per FLOP.
// LDS halo stored channel-last [kh][col][c] (padded) so A-frags are b128 loads.
template <int KS, int CIN>
__global__ __launch_bounds__(256)
void k_conv_wmma(const __bf16* __restrict__ x, const __bf16* __restrict__ wgt,
                 const float* __restrict__ bias, float* __restrict__ of32,
                 __bf16* __restrict__ obf, int Cout, int out_ctot,
                 int out_cbase, int relu) {
  constexpr int PAD = KS / 2;
  constexpr int PW = 32 + 2 * PAD;
  constexpr int KK = KS * KS;
  constexpr int K = CIN * KK;
  constexpr int CP = CIN + 8;                       // padded channel stride
  __shared__ __align__(16) __bf16 smem[KS * PW * CP];
  const int tid = threadIdx.x;
  const int wt = blockIdx.x % WT2_;
  const int nh = blockIdx.x / WT2_;
  const int hh = nh % H_, nn = nh / H_;
  const int w0 = wt * 32;
  // stage input halo: reads coalesced along col, write channel-last
  for (int idx = tid; idx < CIN * KS * PW; idx += 256) {
    int c = idx / (KS * PW);
    int r2 = idx % (KS * PW);
    int kh = r2 / PW, col = r2 % PW;
    int gx = w0 + col - PAD, gy = hh + kh - PAD;
    __bf16 v = (__bf16)0.0f;
    if (gx >= 0 && gx < W_ && gy >= 0 && gy < H_)
      v = x[(((size_t)nn * CIN + c) * H_ + gy) * W_ + gx];
    smem[(kh * PW + col) * CP + c] = v;
  }
  __syncthreads();

  const int lane = tid & 31, wave = tid >> 5;
  const int nbase = blockIdx.y * 128 + wave * 16;
  const int m = lane & 15, agrp = (lane >> 4) * 8;
  const int nl = lane & 15, bgrp = (lane >> 4) * 8;
  const __bf16* wrow = wgt + (size_t)(nbase + nl) * K;
  v8f acc0 = {}, acc1 = {};
#pragma unroll 4
  for (int kb = 0; kb < K; kb += 32) {
    int k0 = kb + agrp;
    int tap0 = k0 / CIN, c0 = k0 % CIN;             // shifts/masks (CIN pow2)
    const __bf16* pa0 = smem + ((tap0 / KS) * PW + m + tap0 % KS) * CP + c0;
    int k1 = kb + 16 + agrp;
    int tap1 = k1 / CIN, c1 = k1 % CIN;
    const __bf16* pa1 = smem + ((tap1 / KS) * PW + m + tap1 % KS) * CP + c1;
    Frag a0, a1, b;
    a0.u[0] = *(const v4ui*)pa0;
    a0.u[1] = *(const v4ui*)pa1;
    a1.u[0] = *(const v4ui*)(pa0 + 16 * CP);        // spatial sub-tile +16 cols
    a1.u[1] = *(const v4ui*)(pa1 + 16 * CP);
    b.u[0] = *(const v4ui*)(wrow + kb + bgrp);
    b.u[1] = *(const v4ui*)(wrow + kb + bgrp + 16);
    acc0 = __builtin_amdgcn_wmma_f32_16x16x32_bf16(false, a0.v, false, b.v, (short)0, acc0,
                                                   false, false);
    acc1 = __builtin_amdgcn_wmma_f32_16x16x32_bf16(false, a1.v, false, b.v, (short)0, acc1,
                                                   false, false);
  }
  const int co = nbase + nl;
  if (co < Cout) {
    float bv = bias ? bias[co] : 0.0f;
    const int mrow = (lane >> 4) * 8;
    size_t ob = (((size_t)nn * out_ctot + out_cbase + co) * H_ + hh) * W_ + w0 + mrow;
#pragma unroll
    for (int t = 0; t < 2; ++t) {
      const v8f& acc = t ? acc1 : acc0;
      float vv[8];
#pragma unroll
      for (int rr = 0; rr < 8; ++rr) {
        float v = acc[rr] + bv;
        vv[rr] = relu ? fmaxf(v, 0.0f) : v;
      }
      size_t o = ob + t * 16;
      if (of32) {
        v4f f0 = {vv[0], vv[1], vv[2], vv[3]};
        v4f f1 = {vv[4], vv[5], vv[6], vv[7]};
        *(v4f*)(of32 + o) = f0;
        *(v4f*)(of32 + o + 4) = f1;
      }
      if (obf) {
        v4ui pk = {pack2bf(vv[0], vv[1]), pack2bf(vv[2], vv[3]),
                   pack2bf(vv[4], vv[5]), pack2bf(vv[6], vv[7])};
        *(v4ui*)(obf + o) = pk;
      }
    }
  }
}

// ---------------------------------------------------------------- scalar heads
__global__ void k_conv7_f1(const float* __restrict__ flow, const float* __restrict__ w,
                           const float* __restrict__ b, __bf16* __restrict__ out) {
  size_t i = (size_t)blockIdx.x * 256 + threadIdx.x;
  size_t total = (size_t)N_ * 128 * HW_;
  if (i >= total) return;
  int ww = (int)(i % W_);
  int hh = (int)((i / W_) % H_);
  int co = (int)((i / HW_) % 128);
  int nn = (int)(i / ((size_t)128 * HW_));
  float acc = b[co];
  for (int kh = 0; kh < 7; ++kh) {
    int gy = hh + kh - 3; if (gy < 0 || gy >= H_) continue;
    for (int kw = 0; kw < 7; ++kw) {
      int gx = ww + kw - 3; if (gx < 0 || gx >= W_) continue;
      acc += w[co * 49 + kh * 7 + kw] * flow[(size_t)nn * HW_ + gy * W_ + gx];
    }
  }
  out[i] = (__bf16)fmaxf(acc, 0.0f);
}

__global__ void k_fh2(const __bf16* __restrict__ d1, const float* __restrict__ w,
                      const float* __restrict__ b, float* __restrict__ flow) {
  size_t i = (size_t)blockIdx.x * 256 + threadIdx.x;
  if (i >= NHW_) return;
  int ww = (int)(i % W_);
  int hh = (int)((i / W_) % H_);
  int nn = (int)(i / HW_);
  float acc = b[0];
  for (int c = 0; c < 128; ++c)
    for (int kh = 0; kh < 3; ++kh) {
      int gy = hh + kh - 1; if (gy < 0 || gy >= H_) continue;
      for (int kw = 0; kw < 3; ++kw) {
        int gx = ww + kw - 1; if (gx < 0 || gx >= W_) continue;
        acc += (float)d1[(((size_t)nn * 128 + c) * H_ + gy) * W_ + gx] *
               w[(c * 3 + kh) * 3 + kw];
      }
    }
  flow[i] += acc;
}

__global__ void k_set_flow_ch(const float* __restrict__ flow, __bf16* __restrict__ mo) {
  size_t i = (size_t)blockIdx.x * 256 + threadIdx.x;
  if (i >= NHW_) return;
  int hw = (int)(i % HW_);
  int nn = (int)(i / HW_);
  mo[((size_t)nn * 128 + 127) * HW_ + hw] = (__bf16)flow[i];
}

__global__ void k_gate(const float* __restrict__ a, const float* __restrict__ b,
                       float* __restrict__ g, size_t n) {
  size_t i = (size_t)blockIdx.x * 256 + threadIdx.x;
  if (i < n) g[i] = 1.0f / (1.0f + expf(-(a[i] + b[i])));
}

__global__ void k_gru_update(const float* __restrict__ xq, const float* __restrict__ hq,
                             const float* __restrict__ r, const float* __restrict__ z,
                             float* __restrict__ hf, __bf16* __restrict__ hb, size_t n) {
  size_t i = (size_t)blockIdx.x * 256 + threadIdx.x;
  if (i >= n) return;
  float q = tanhf(xq[i] + r[i] * hq[i]);
  float zz = z[i];
  float hn = (1.0f - zz) * q + zz * hf[i];
  hf[i] = hn;
  hb[i] = (__bf16)hn;
}

__global__ void k_upsample(const float* __restrict__ flow, const __bf16* __restrict__ mask,
                           float* __restrict__ out) {
  size_t i = (size_t)blockIdx.x * 256 + threadIdx.x;
  size_t total = (size_t)N_ * (8 * H_) * (8 * W_);
  if (i >= total) return;
  int ow = (int)(i % (8 * W_));
  int oh = (int)((i / (8 * W_)) % (8 * H_));
  int nn = (int)(i / ((size_t)(8 * W_) * (8 * H_)));
  int aa = oh / H_, hh = oh % H_;
  int bb = ow / W_, ww = ow % W_;
  int ds2 = aa * 8 + bb;
  float lg[9], mx = -1e30f;
#pragma unroll
  for (int j = 0; j < 9; ++j) {
    lg[j] = 0.25f * (float)mask[(((size_t)nn * 576 + j * 64 + ds2) * H_ + hh) * W_ + ww];
    mx = fmaxf(mx, lg[j]);
  }
  float s = 0.0f;
#pragma unroll
  for (int j = 0; j < 9; ++j) { lg[j] = expf(lg[j] - mx); s += lg[j]; }
  float acc = 0.0f;
#pragma unroll
  for (int j = 0; j < 9; ++j) {
    int gy = hh + j / 3 - 1, gx = ww + j % 3 - 1;
    float fv = (gy >= 0 && gy < H_ && gx >= 0 && gx < W_)
                   ? 8.0f * flow[(size_t)nn * HW_ + gy * W_ + gx] : 0.0f;
    acc += lg[j] * fv;
  }
  out[i] = acc / s;
}

// ---------------------------------------------------------------- launch
extern "C" void kernel_launch(void* const* d_in, const int* in_sizes, int n_in,
                              void* d_out, int out_size, void* d_ws, size_t ws_size,
                              hipStream_t stream) {
  const float* l_fmap = (const float*)d_in[0];
  const float* r_fmap = (const float*)d_in[1];
  const float* h_init = (const float*)d_in[2];
  const float* P[30];
  for (int i = 0; i < 30; ++i) P[i] = (const float*)d_in[3 + i];

  char* p = (char*)d_ws;
  auto alloc = [&](size_t bytes) -> void* {
    void* r = (void*)p; p += (bytes + 255) & ~(size_t)255; return r;
  };
  __bf16* l_bf   = (__bf16*)alloc(NCHW_ * 2);
  __bf16* r_bf   = (__bf16*)alloc(NCHW_ * 2);
  float* pyr0    = (float*)alloc(NHW_ * 160 * 4);
  float* pyr1    = (float*)alloc(NHW_ * 80 * 4);
  float* pyr2    = (float*)alloc(NHW_ * 40 * 4);
  float* pyr3    = (float*)alloc(NHW_ * 20 * 4);
  __bf16* corrf  = (__bf16*)alloc((size_t)N_ * 64 * HW_ * 2);
  __bf16* tmpact = (__bf16*)alloc(NCHW_ * 2);
  __bf16* cat    = (__bf16*)alloc(2 * NCHW_ * 2);
  __bf16* mo     = (__bf16*)alloc(NCHW_ * 2);
  __bf16* h_bf   = (__bf16*)alloc(NCHW_ * 2);
  float* h_f32   = (float*)alloc(NCHW_ * 4);
  float* tmpA    = (float*)alloc(NCHW_ * 4);
  float* tmpB    = (float*)alloc(NCHW_ * 4);
  float* zbuf    = (float*)alloc(NCHW_ * 4);
  float* rbuf    = (float*)alloc(NCHW_ * 4);
  float* flow    = (float*)alloc(NHW_ * 4);
  __bf16* maskb  = (__bf16*)alloc((size_t)N_ * 576 * HW_ * 2);
  __bf16* w_c1   = (__bf16*)alloc((size_t)128 * 64 * 2);
  __bf16* w_c2   = (__bf16*)alloc((size_t)128 * 1152 * 2);
  __bf16* w_f2   = (__bf16*)alloc((size_t)128 * 1152 * 2);
  __bf16* w_mc   = (__bf16*)alloc((size_t)128 * 2304 * 2);
  __bf16* w_g[6];
  for (int i = 0; i < 6; ++i) w_g[i] = (__bf16*)alloc((size_t)128 * 1152 * 2);
  __bf16* w_fh1  = (__bf16*)alloc((size_t)128 * 1152 * 2);
  __bf16* w_mh1  = (__bf16*)alloc((size_t)128 * 1152 * 2);
  __bf16* w_mh2  = (__bf16*)alloc((size_t)640 * 128 * 2);

  auto blocks = [](size_t n) { return (unsigned)((n + 255) / 256); };
  auto pack = [&](const float* src, __bf16* dst, int Cout, int CoP, int CinS, int CinD, int KK) {
    k_pack_w<<<blocks((size_t)CoP * CinD * KK), 256, 0, stream>>>(src, dst, Cout, CoP,
                                                                  CinS, CinD, KK);
  };
  pack(P[0],  w_c1, 128, 128, 36, 64, 1);
  pack(P[2],  w_c2, 128, 128, 128, 128, 9);
  pack(P[6],  w_f2, 128, 128, 128, 128, 9);
  pack(P[8],  w_mc, 127, 128, 256, 256, 9);
  for (int i = 0; i < 6; ++i) pack(P[10 + 2 * i], w_g[i], 128, 128, 128, 128, 9);
  pack(P[22], w_fh1, 128, 128, 128, 128, 9);
  pack(P[26], w_mh1, 128, 128, 128, 128, 9);
  pack(P[28], w_mh2, 576, 640, 128, 128, 1);

  k_f32_to_bf16<<<blocks(NCHW_), 256, 0, stream>>>(l_fmap, l_bf, NCHW_);
  k_f32_to_bf16<<<blocks(NCHW_), 256, 0, stream>>>(r_fmap, r_bf, NCHW_);
  k_hprep<<<blocks(NCHW_), 256, 0, stream>>>(h_init, h_f32, h_bf, NCHW_);
  k_zero_f32<<<blocks(NHW_), 256, 0, stream>>>(flow, NHW_);

  k_corr_wmma<<<N_ * H_ * WT_, 320, 0, stream>>>(l_bf, r_bf, pyr0);
  k_pool_half<<<blocks(NHW_ * 80), 256, 0, stream>>>(pyr0, pyr1, 160, NHW_);
  k_pool_half<<<blocks(NHW_ * 40), 256, 0, stream>>>(pyr1, pyr2, 80, NHW_);
  k_pool_half<<<blocks(NHW_ * 20), 256, 0, stream>>>(pyr2, pyr3, 40, NHW_);

  auto conv3_128 = [&](const __bf16* x, const __bf16* w, const float* b, float* of,
                       __bf16* ob, int Cout, int ctot, int cbase, int relu) {
    dim3 g(N_ * H_ * WT2_, (Cout + 127) / 128);
    k_conv_wmma<3, 128><<<g, 256, 0, stream>>>(x, w, b, of, ob, Cout, ctot, cbase, relu);
  };
  auto conv3_256 = [&](const __bf16* x, const __bf16* w, const float* b, float* of,
                       __bf16* ob, int Cout, int ctot, int cbase, int relu) {
    dim3 g(N_ * H_ * WT2_, (Cout + 127) / 128);
    k_conv_wmma<3, 256><<<g, 256, 0, stream>>>(x, w, b, of, ob, Cout, ctot, cbase, relu);
  };
  auto conv1_64 = [&](const __bf16* x, const __bf16* w, const float* b, float* of,
                      __bf16* ob, int Cout, int ctot, int cbase, int relu) {
    dim3 g(N_ * H_ * WT2_, (Cout + 127) / 128);
    k_conv_wmma<1, 64><<<g, 256, 0, stream>>>(x, w, b, of, ob, Cout, ctot, cbase, relu);
  };
  auto conv1_128 = [&](const __bf16* x, const __bf16* w, const float* b, float* of,
                       __bf16* ob, int Cout, int ctot, int cbase, int relu) {
    dim3 g(N_ * H_ * WT2_, (Cout + 127) / 128);
    k_conv_wmma<1, 128><<<g, 256, 0, stream>>>(x, w, b, of, ob, Cout, ctot, cbase, relu);
  };

  for (int it = 0; it < 6; ++it) {
    k_sample<<<blocks((size_t)N_ * 64 * HW_), 256, 0, stream>>>(pyr0, pyr1, pyr2, pyr3,
                                                                flow, corrf);
    // motion head
    conv1_64(corrf, w_c1, P[1], nullptr, tmpact, 128, 128, 0, 1);           // cf1
    conv3_128(tmpact, w_c2, P[3], nullptr, cat, 128, 256, 128, 1);          // cf2 -> cat hi
    k_conv7_f1<<<blocks((size_t)N_ * 128 * HW_), 256, 0, stream>>>(flow, P[4], P[5], tmpact);
    conv3_128(tmpact, w_f2, P[7], nullptr, cat, 128, 256, 0, 1);            // ff2 -> cat lo
    conv3_256(cat, w_mc, P[9], nullptr, mo, 127, 128, 0, 1);                // mc
    k_set_flow_ch<<<blocks(NHW_), 256, 0, stream>>>(flow, mo);
    // GRU
    conv3_128(mo,   w_g[0], P[11], tmpA, nullptr, 128, 128, 0, 0);          // xz
    conv3_128(h_bf, w_g[1], P[13], tmpB, nullptr, 128, 128, 0, 0);          // hz
    k_gate<<<blocks(NCHW_), 256, 0, stream>>>(tmpA, tmpB, zbuf, NCHW_);
    conv3_128(mo,   w_g[2], P[15], tmpA, nullptr, 128, 128, 0, 0);          // xr
    conv3_128(h_bf, w_g[3], P[17], tmpB, nullptr, 128, 128, 0, 0);          // hr
    k_gate<<<blocks(NCHW_), 256, 0, stream>>>(tmpA, tmpB, rbuf, NCHW_);
    conv3_128(mo,   w_g[4], P[19], tmpA, nullptr, 128, 128, 0, 0);          // xq
    conv3_128(h_bf, w_g[5], P[21], tmpB, nullptr, 128, 128, 0, 0);          // hq
    k_gru_update<<<blocks(NCHW_), 256, 0, stream>>>(tmpA, tmpB, rbuf, zbuf, h_f32, h_bf,
                                                    NCHW_);
    // flow head
    conv3_128(h_bf, w_fh1, P[23], nullptr, tmpact, 128, 128, 0, 1);
    k_fh2<<<blocks(NHW_), 256, 0, stream>>>(tmpact, P[24], P[25], flow);
    // mask head only needed on the final iteration
    if (it == 5) {
      conv3_128(h_bf, w_mh1, P[27], nullptr, tmpact, 128, 128, 0, 1);
      conv1_128(tmpact, w_mh2, P[29], nullptr, maskb, 576, 576, 0, 0);
    }
  }
  k_upsample<<<blocks((size_t)N_ * 768 * 1280), 256, 0, stream>>>(flow, maskb, (float*)d_out);
  (void)in_sizes; (void)n_in; (void)out_size; (void)ws_size;
}
